// NeighborAttention_5600637354122
// MI455X (gfx1250) — matrix operand order repaired and checked by reading, not compiled, not run on hardware
//
#include <hip/hip_runtime.h>
#include <hip/hip_bf16.h>

// ---------------------------------------------------------------------------
// NeighborAttention on MI455X (gfx1250, wave32, WMMA bf16 16x16x32)
//
//   Q = xWq+bq; K = xWk+bk; V = xWv+bv
//   s = rowsum(adj * (Q K^T))  ==  per-row dot(Q_i, (adj @ K)_i)
//   alpha = softmax(s); out = alpha[:,None] * V
//
// Big GEMM (adj @ K) is double-buffered through LDS with
// GLOBAL_LOAD_ASYNC_TO_LDS_B128 (ASYNCcnt) when the toolchain exposes it.
// ---------------------------------------------------------------------------

#define NN 8192
#define DD 256

typedef __attribute__((ext_vector_type(16))) __bf16 v16bf;
typedef __attribute__((ext_vector_type(8)))  float  v8f;
typedef __attribute__((ext_vector_type(4)))  float  f32x4;
typedef __attribute__((ext_vector_type(4)))  unsigned int u32x4;

// GCC-style int4 vector: matches the async-to-LDS builtin's parameter type
typedef int v4i __attribute__((vector_size(4 * sizeof(int))));
typedef __attribute__((address_space(1))) v4i* g_v4i_p;   // global
typedef __attribute__((address_space(3))) v4i* l_v4i_p;   // LDS

#if __has_builtin(__builtin_amdgcn_global_load_async_to_lds_b128) && \
    __has_builtin(__builtin_amdgcn_s_wait_asynccnt)
#define HAS_ASYNC 1
#else
#define HAS_ASYNC 0
#endif

union Frag16 {            // one WMMA 16-bit A/B operand (8 VGPRs / lane)
  v16bf v;
  u32x4 q[2];
};

__device__ __forceinline__ unsigned short f2bf(float f) {
  unsigned int b = __float_as_uint(f);
  b += 0x7FFFu + ((b >> 16) & 1u);   // round-to-nearest-even
  return (unsigned short)(b >> 16);
}
__device__ __forceinline__ unsigned int pk2(float x, float y) {
  return (unsigned int)f2bf(x) | ((unsigned int)f2bf(y) << 16);
}

// ---------------- elementwise f32 -> bf16 -----------------------------------
__global__ void k_f32_to_bf16(const float* __restrict__ in,
                              unsigned short* __restrict__ out, int n) {
  int i = blockIdx.x * blockDim.x + threadIdx.x;
  if (i < n) out[i] = f2bf(in[i]);
}

// ---------------- W (256x256 f32) -> W^T bf16 (n-major) ---------------------
__global__ void k_transpose_bf16(const float* __restrict__ W,
                                 unsigned short* __restrict__ Wt) {
  int t = blockIdx.x * blockDim.x + threadIdx.x;   // 65536 threads
  int n = t >> 8, k = t & 255;
  Wt[n * DD + k] = f2bf(W[k * DD + n]);
}

// ---------------- projection GEMM: out = x_bf16 @ W + b ---------------------
// A: x bf16 row-major (8192x256); Bt: W^T bf16 (n-major, 256x256)
// block = 128 threads (4 waves); tile 64(M) x 64(N); wave w owns mtile w.
// mode 0: write f32 row-major (Q, V). mode 1: write bf16 transposed (Kt).
__global__ void k_proj(const unsigned short* __restrict__ A,
                       const unsigned short* __restrict__ Bt,
                       const float* __restrict__ bias,
                       float* __restrict__ outF,
                       unsigned short* __restrict__ outKt,
                       int mode) {
  const int m0   = blockIdx.x * 64;
  const int n0   = blockIdx.y * 64;
  const int tid  = threadIdx.x;
  const int wave = tid >> 5;
  const int lane = tid & 31;
  const int l16  = lane & 15;
  const int hi   = lane >> 4;
  const int mrow = m0 + wave * 16 + l16;

  v8f acc[4] = {};
  for (int k0 = 0; k0 < DD; k0 += 32) {
    Frag16 a;
    // A lane layout (16x32 bf16): lanes<16 hold K {0..7,16..23}, lanes>=16 {8..15,24..31}
    const u32x4* pa = (const u32x4*)(A + (size_t)mrow * DD + k0 + hi * 8);
    a.q[0] = pa[0];
    a.q[1] = pa[2];
#pragma unroll
    for (int t = 0; t < 4; ++t) {
      Frag16 b;
      // B lane layout (32x16 bf16): column = lane%16, lanes<16 hold K {0..7,16..23}, lanes>=16 {8..15,24..31}
      const u32x4* pb = (const u32x4*)(Bt + (size_t)(n0 + t * 16 + l16) * DD + k0 + hi * 8);
      b.q[0] = pb[0];
      b.q[1] = pb[2];
      acc[t] = __builtin_amdgcn_wmma_f32_16x16x32_bf16(
          false, a.v, false, b.v, (short)0, acc[t], false, false);
    }
  }
#pragma unroll
  for (int t = 0; t < 4; ++t) {
    int n = n0 + t * 16 + l16;
    float bn = bias[n];
#pragma unroll
    for (int r = 0; r < 8; ++r) {
      int row = m0 + wave * 16 + r + hi * 8;   // C/D layout: VGPR r -> M=r (+8 for hi lanes)
      float val = acc[t][r] + bn;
      if (mode == 0) outF[(size_t)row * DD + n] = val;
      else           outKt[(size_t)n * NN + row] = f2bf(val);
    }
  }
}

// ---------------- big GEMM + fused dot: s = diag(Q (adj K)^T) ---------------
// block = 256 threads (8 waves); tile 64 rows x 256 cols (full width -> adj read
// exactly once from HBM). wave w: mtile = w&3, n-range = (w>>2)*128 .. +127.
// LDS holds raw f32 adj tiles (double buffered, K=64 per stage), copied with
// async-to-LDS when available; waves convert to bf16 while building A frags.
#define KT    64                 // k-depth per stage (2 WMMA k-steps)
#define ASTR  68                 // f32 row stride in LDS (64 + 4 pad, 272B)
#define NSTAGE (NN / KT)         // 128

__global__ void k_adj_gemm_s(const float* __restrict__ adj,          // 8192x8192 f32
                             const unsigned short* __restrict__ Kt,  // 256x8192 bf16
                             const float* __restrict__ Q,            // 8192x256 f32
                             float* __restrict__ s) {
  __shared__ __align__(16) float ldsA[2][64 * ASTR];   // 2 x 17 KB
  __shared__ float sred[8][16];

  const int m0    = blockIdx.x * 64;
  const int tid   = threadIdx.x;
  const int wave  = tid >> 5;
  const int lane  = tid & 31;
  const int l16   = lane & 15;
  const int hi    = lane >> 4;
  const int mtile = wave & 3;
  const int nbase = (wave >> 2) * 128;

  v8f acc[8] = {};

  // ---- stage: copy adj[m0..m0+63][k0..k0+KT-1] (f32) into LDS buffer b ----
  // 64 rows x 16 float4 chunks = 1024 chunks; 4 per thread.
  auto stage = [&](int b, int k0) {
#pragma unroll
    for (int j = 0; j < 4; ++j) {
      int slot = tid + j * 256;
      int row  = slot >> 4;
      int c4   = slot & 15;
      const float* g = adj + (size_t)(m0 + row) * NN + k0 + c4 * 4;
      float*       l = &ldsA[b][row * ASTR + c4 * 4];
#if HAS_ASYNC
      __builtin_amdgcn_global_load_async_to_lds_b128((g_v4i_p)g, (l_v4i_p)l, 0, 0);
#else
      *(f32x4*)l = *(const f32x4*)g;
#endif
    }
  };

  // ---- compute: 2 k-steps of 8 WMMAs from LDS buffer b --------------------
  auto compute = [&](int b, int k0) {
#pragma unroll
    for (int kk = 0; kk < KT; kk += 32) {
      Frag16 a;
      const float* base = &ldsA[b][(mtile * 16 + l16) * ASTR + kk + hi * 8];
      f32x4 a0 = ((const f32x4*)base)[0];        // k: hi*8   .. +3
      f32x4 a1 = ((const f32x4*)base)[1];        // k: hi*8+4 .. +7
      f32x4 a2 = ((const f32x4*)(base + 16))[0]; // k: hi*8+16 .. +19
      f32x4 a3 = ((const f32x4*)(base + 16))[1]; // k: hi*8+20 .. +23
      a.q[0][0] = pk2(a0[0], a0[1]);  a.q[0][1] = pk2(a0[2], a0[3]);
      a.q[0][2] = pk2(a1[0], a1[1]);  a.q[0][3] = pk2(a1[2], a1[3]);
      a.q[1][0] = pk2(a2[0], a2[1]);  a.q[1][1] = pk2(a2[2], a2[3]);
      a.q[1][2] = pk2(a3[0], a3[1]);  a.q[1][3] = pk2(a3[2], a3[3]);
#pragma unroll
      for (int t = 0; t < 8; ++t) {
        Frag16 bf;
        const u32x4* pb = (const u32x4*)(Kt + (size_t)(nbase + t * 16 + l16) * NN +
                                         k0 + kk + hi * 8);
        bf.q[0] = pb[0];
        bf.q[1] = pb[2];
        acc[t] = __builtin_amdgcn_wmma_f32_16x16x32_bf16(
            false, a.v, false, bf.v, (short)0, acc[t], false, false);
      }
    }
  };

  // ---- double-buffered pipeline -------------------------------------------
  stage(0, 0);
  for (int ks = 0; ks < NSTAGE - 1; ++ks) {
    __syncthreads();                       // everyone done reading buf[(ks+1)&1]
    stage((ks + 1) & 1, (ks + 1) * KT);    // prefetch next tile (async)
#if HAS_ASYNC
    __builtin_amdgcn_s_wait_asynccnt(4);   // current tile's 4 copies landed
#endif
    __syncthreads();                       // all waves' copies visible
    compute(ks & 1, ks * KT);
  }
#if HAS_ASYNC
  __builtin_amdgcn_s_wait_asynccnt(0);
#endif
  __syncthreads();
  compute((NSTAGE - 1) & 1, (NSTAGE - 1) * KT);

  // ---- fused epilogue: p[r] = sum over this wave's 128 cols of m*Q --------
  float p[8];
#pragma unroll
  for (int r = 0; r < 8; ++r) {
    int row = m0 + mtile * 16 + r + hi * 8;
    float d = 0.f;
#pragma unroll
    for (int t = 0; t < 8; ++t) {
      int n = nbase + t * 16 + l16;
      d += acc[t][r] * Q[(size_t)row * DD + n];
    }
    p[r] = d;
  }
#pragma unroll
  for (int r = 0; r < 8; ++r)
    for (int off = 1; off < 16; off <<= 1)
      p[r] += __shfl_xor(p[r], off, 32);

  if (l16 == 0) {
#pragma unroll
    for (int r = 0; r < 8; ++r) sred[wave][hi * 8 + r] = p[r];
  }
  __syncthreads();
  if (tid < 64) {
    int mt = tid >> 4, rr = tid & 15;
    s[m0 + tid] = sred[mt][rr] + sred[mt + 4][rr];   // combine the two n-halves
  }
}

// ---------------- softmax over 8192 values (single block) -------------------
__global__ void k_softmax(const float* __restrict__ s, float* __restrict__ alpha, int n) {
  __shared__ float red[1024];
  int tid = threadIdx.x;
  float m = -3.4e38f;
  for (int i = tid; i < n; i += 1024) m = fmaxf(m, s[i]);
  red[tid] = m; __syncthreads();
  for (int off = 512; off > 0; off >>= 1) {
    if (tid < off) red[tid] = fmaxf(red[tid], red[tid + off]);
    __syncthreads();
  }
  float mx = red[0]; __syncthreads();
  float sum = 0.f;
  for (int i = tid; i < n; i += 1024) sum += __expf(s[i] - mx);
  red[tid] = sum; __syncthreads();
  for (int off = 512; off > 0; off >>= 1) {
    if (tid < off) red[tid] += red[tid + off];
    __syncthreads();
  }
  float inv = 1.0f / red[0];
  for (int i = tid; i < n; i += 1024) alpha[i] = __expf(s[i] - mx) * inv;
}

// ---------------- out = alpha[i] * V ----------------------------------------
__global__ void k_scale(const float* __restrict__ V, const float* __restrict__ alpha,
                        float* __restrict__ out) {
  int i = blockIdx.x * blockDim.x + threadIdx.x;
  out[i] = alpha[i >> 8] * V[i];
}

// ---------------------------------------------------------------------------
extern "C" void kernel_launch(void* const* d_in, const int* in_sizes, int n_in,
                              void* d_out, int out_size, void* d_ws, size_t ws_size,
                              hipStream_t stream) {
  const float* x   = (const float*)d_in[0];
  const float* adj = (const float*)d_in[1];
  const float* Wq  = (const float*)d_in[2];
  const float* bq  = (const float*)d_in[3];
  const float* Wk  = (const float*)d_in[4];
  const float* bk  = (const float*)d_in[5];
  const float* Wv  = (const float*)d_in[6];
  const float* bv  = (const float*)d_in[7];

  float* out   = (float*)d_out;                 // 8192*256 f32
  float* alpha = out + (size_t)NN * DD;         // + 8192 f32

  // workspace layout (~24.5 MiB)
  char* ws = (char*)d_ws;
  unsigned short* xbf = (unsigned short*)ws;                       // 4 MiB
  unsigned short* Wtq = (unsigned short*)(ws + ((size_t)NN * DD * 2));
  unsigned short* Wtk = Wtq + (size_t)DD * DD;
  unsigned short* Wtv = Wtk + (size_t)DD * DD;
  float* Qm = (float*)(Wtv + (size_t)DD * DD);                     // 8 MiB
  float* Vm = Qm + (size_t)NN * DD;                                // 8 MiB
  unsigned short* Ktb = (unsigned short*)(Vm + (size_t)NN * DD);   // 4 MiB (256 x 8192)
  float* sbuf = (float*)(Ktb + (size_t)DD * NN);                   // 32 KiB

  k_f32_to_bf16<<<(NN * DD) / 256, 256, 0, stream>>>(x, xbf, NN * DD);
  k_transpose_bf16<<<(DD * DD) / 256, 256, 0, stream>>>(Wq, Wtq);
  k_transpose_bf16<<<(DD * DD) / 256, 256, 0, stream>>>(Wk, Wtk);
  k_transpose_bf16<<<(DD * DD) / 256, 256, 0, stream>>>(Wv, Wtv);

  dim3 gp(NN / 64, DD / 64);
  k_proj<<<gp, 128, 0, stream>>>(xbf, Wtq, bq, Qm, (unsigned short*)nullptr, 0);
  k_proj<<<gp, 128, 0, stream>>>(xbf, Wtv, bv, Vm, (unsigned short*)nullptr, 0);
  k_proj<<<gp, 128, 0, stream>>>(xbf, Wtk, bk, (float*)nullptr, Ktb, 1);

  k_adj_gemm_s<<<NN / 64, 256, 0, stream>>>(adj, Ktb, Qm, sbuf);
  k_softmax<<<1, 1024, 0, stream>>>(sbuf, alpha, NN);
  k_scale<<<(NN * DD) / 256, 256, 0, stream>>>(Vm, alpha, out);
}
